// InfoNCE_22806276342428
// MI455X (gfx1250) — matrix operand order repaired and checked by reading, not compiled, not run on hardware
//
#include <hip/hip_runtime.h>
#include <hip/hip_bf16.h>
#include <math.h>

typedef __attribute__((ext_vector_type(16))) __bf16 v16bf;
typedef __attribute__((ext_vector_type(8)))  float  v8f;
typedef __attribute__((ext_vector_type(4)))  int    v4i;

#define NN 1536
#define HID 64
#define DX 192
#define W1STRIDE 384

#define GAS __attribute__((address_space(1)))
#define LAS __attribute__((address_space(3)))

#if __has_builtin(__builtin_amdgcn_global_load_async_to_lds_b128) && \
    __has_builtin(__builtin_amdgcn_s_wait_asynccnt)
#define USE_ASYNC_LDS 1
#else
#define USE_ASYNC_LDS 0
#endif

// ---------------------------------------------------------------------------
// Kernel 1: hx = x @ W1x^T ; hy = y @ W1y^T + b1   via v_wmma_f32_16x16x32_bf16
// One wave (32 threads) per 16-row tile. K = 192 = 6 steps of 32.
// ---------------------------------------------------------------------------
__global__ __launch_bounds__(32) void hxhy_wmma(
    const float* __restrict__ x, const float* __restrict__ y,
    const float* __restrict__ W1, const float* __restrict__ b1,
    float* __restrict__ hx, float* __restrict__ hyb)
{
    const int tile  = blockIdx.x;          // 0..95
    const int lane  = threadIdx.x & 31;
    const int m     = lane & 15;           // row within tile (A) / col N (B,C)
    const int khalf = lane >> 4;           // 0 or 1
    const int row   = tile * 16 + m;

    v8f accx[4] = {}; v8f accy[4] = {};

    const float* xr = x + row * DX;
    const float* yr = y + row * DX;

    for (int kb = 0; kb < 6; ++kb) {
        // A layout (16-bit 16x32): lanes 0-15 hold K 0..15; lanes 16-31 hold K 16..31 (matches B)
        const int k0 = kb * 32 + khalf * 16;
        const int k1 = kb * 32 + khalf * 16 + 8;
        v16bf ax, ay;
        #pragma unroll
        for (int t = 0; t < 8; ++t) {
            ax[t]     = (__bf16)xr[k0 + t];
            ax[8 + t] = (__bf16)xr[k1 + t];
            ay[t]     = (__bf16)yr[k0 + t];
            ay[8 + t] = (__bf16)yr[k1 + t];
        }
        // B layout (32x16): lanes 0-15 hold K 0..15, lanes 16-31 hold K 16..31; N = lane&15
        const int kbB = kb * 32 + khalf * 16;
        #pragma unroll
        for (int c = 0; c < 4; ++c) {
            const float* wrow = W1 + (c * 16 + m) * W1STRIDE;   // B[k][n] = W1[c*16+n][k]
            v16bf bx, by;
            #pragma unroll
            for (int t = 0; t < 16; ++t) {
                bx[t] = (__bf16)wrow[kbB + t];          // W1x part (cols 0..191)
                by[t] = (__bf16)wrow[DX + kbB + t];     // W1y part (cols 192..383)
            }
            accx[c] = __builtin_amdgcn_wmma_f32_16x16x32_bf16(
                          false, ax, false, bx, (short)0, accx[c], false, false);
            accy[c] = __builtin_amdgcn_wmma_f32_16x16x32_bf16(
                          false, ay, false, by, (short)0, accy[c], false, false);
        }
    }

    // C/D layout: lanes 0-15 -> N=lane, M=r ; lanes 16-31 -> N=lane-16, M=r+8
    const int n  = lane & 15;
    const int mb = khalf * 8;
    #pragma unroll
    for (int c = 0; c < 4; ++c) {
        const float bb = b1[c * 16 + n];
        #pragma unroll
        for (int r = 0; r < 8; ++r) {
            const int grow = tile * 16 + mb + r;
            hx[grow * HID + c * 16 + n]  = accx[c][r];
            hyb[grow * HID + c * 16 + n] = accy[c][r] + bb;   // fold b1 into hy
        }
    }
}

// ---------------------------------------------------------------------------
// Kernel 2: T1[i,j] = softplus(sum_h W2[h]*relu(hyb[i,h]+hx[j,h]) + b2)
// streaming logsumexp over j + diagonal capture.
// 32 i's per block (2 per thread), j chunks of 64, double-buffered async LDS.
// ---------------------------------------------------------------------------
#define TJ 64
#define ITILE 32
#define HXPAD 68                 // row stride: 272B, 16B aligned, conflict-free
#define NCHUNK (NN / TJ)

__global__ __launch_bounds__(256) void pairwise_lse(
    const float* __restrict__ hx, const float* __restrict__ hyb,
    const float* __restrict__ W2, const float* __restrict__ b2p,
    float* __restrict__ diag, float* __restrict__ lse)
{
    __shared__ float hy_s[ITILE][65];
    __shared__ float hx_s[2][TJ][HXPAD];
    __shared__ float w2_s[HID];

    const int tid   = threadIdx.x;
    const int g     = tid >> 4;          // 0..15 : i-pair group
    const int j_sub = tid & 15;
    const int i0    = blockIdx.x * ITILE + 2 * g;

    if (tid < HID) w2_s[tid] = W2[tid];
    for (int v = tid; v < ITILE * HID; v += 256)
        hy_s[v >> 6][v & 63] = hyb[blockIdx.x * ITILE * HID + v];

    // each thread moves 4x b128 (16 floats) per 16KB chunk
    auto issue_chunk = [&](int c, int buf) {
        const float* src = hx + c * TJ * HID;
        #pragma unroll
        for (int k = 0; k < 4; ++k) {
            const int f = tid + k * 256;       // float4 index, 0..1023
            const int j  = f >> 4;             // row 0..63
            const int h0 = (f & 15) * 4;       // col 0,4,...,60
#if USE_ASYNC_LDS
            __builtin_amdgcn_global_load_async_to_lds_b128(
                (GAS v4i*)(src + f * 4),
                (LAS v4i*)&hx_s[buf][j][h0], 0, 0);
#else
            *(float4*)&hx_s[buf][j][h0] = *(const float4*)(src + f * 4);
#endif
        }
    };

    const float b2v = b2p[0];
    float mx0 = -__builtin_inff(), sm0 = 0.f, dv0 = 0.f;
    float mx1 = -__builtin_inff(), sm1 = 0.f, dv1 = 0.f;

    issue_chunk(0, 0);

    for (int c = 0; c < NCHUNK; ++c) {
        const int cur = c & 1;
        if (c + 1 < NCHUNK) {
            issue_chunk(c + 1, cur ^ 1);       // prev iter's barrier made buf free
#if USE_ASYNC_LDS
            __builtin_amdgcn_s_wait_asynccnt(4);   // in-order: <=4 left => chunk c done
        } else {
            __builtin_amdgcn_s_wait_asynccnt(0);
#endif
        }
        __syncthreads();                       // chunk c visible to all threads

        float a0[4] = {0.f, 0.f, 0.f, 0.f};
        float a1[4] = {0.f, 0.f, 0.f, 0.f};
        #pragma unroll 8
        for (int h = 0; h < HID; ++h) {
            const float w   = w2_s[h];
            const float hv0 = hy_s[2 * g][h];
            const float hv1 = hy_s[2 * g + 1][h];
            #pragma unroll
            for (int q = 0; q < 4; ++q) {
                const float xv = hx_s[cur][j_sub + q * 16][h];
                a0[q] = fmaf(w, fmaxf(hv0 + xv, 0.f), a0[q]);
                a1[q] = fmaf(w, fmaxf(hv1 + xv, 0.f), a1[q]);
            }
        }

        #pragma unroll
        for (int q = 0; q < 4; ++q) {
            const int jg = c * TJ + j_sub + q * 16;
            {
                const float v = a0[q] + b2v;
                const float t = fmaxf(v, 0.f) + log1pf(__expf(-fabsf(v)));
                const float nm = fmaxf(mx0, t);
                sm0 = sm0 * __expf(mx0 - nm) + __expf(t - nm);
                mx0 = nm;
                if (jg == i0) dv0 = t;
            }
            {
                const float v = a1[q] + b2v;
                const float t = fmaxf(v, 0.f) + log1pf(__expf(-fabsf(v)));
                const float nm = fmaxf(mx1, t);
                sm1 = sm1 * __expf(mx1 - nm) + __expf(t - nm);
                mx1 = nm;
                if (jg == i0 + 1) dv1 = t;
            }
        }
        __syncthreads();    // all reads of hx_s[cur] done before it is re-filled
    }

    // merge the 16 lanes sharing an i-pair (each group is one half of a wave32)
    #pragma unroll
    for (int off = 8; off >= 1; off >>= 1) {
        {
            const float omx = __shfl_xor(mx0, off, 32);
            const float osm = __shfl_xor(sm0, off, 32);
            const float nm  = fmaxf(mx0, omx);
            sm0 = sm0 * __expf(mx0 - nm) + osm * __expf(omx - nm);
            mx0 = nm;
        }
        {
            const float omx = __shfl_xor(mx1, off, 32);
            const float osm = __shfl_xor(sm1, off, 32);
            const float nm  = fmaxf(mx1, omx);
            sm1 = sm1 * __expf(mx1 - nm) + osm * __expf(omx - nm);
            mx1 = nm;
        }
        dv0 += __shfl_xor(dv0, off, 32);
        dv1 += __shfl_xor(dv1, off, 32);
    }
    if (j_sub == 0) {
        lse[i0]      = mx0 + __logf(sm0);
        diag[i0]     = dv0;                 // == T1[i,i] == T0[i]
        lse[i0 + 1]  = mx1 + __logf(sm1);
        diag[i0 + 1] = dv1;
    }
}

// ---------------------------------------------------------------------------
// Kernel 3: loss = -( mean(diag) - (mean(lse) - log N) )
// ---------------------------------------------------------------------------
__global__ __launch_bounds__(256) void finalize(
    const float* __restrict__ diag, const float* __restrict__ lse,
    float* __restrict__ out)
{
    __shared__ float sd[256], sl[256];
    const int tid = threadIdx.x;
    float d = 0.f, l = 0.f;
    for (int i = tid; i < NN; i += 256) { d += diag[i]; l += lse[i]; }
    sd[tid] = d; sl[tid] = l;
    __syncthreads();
    for (int s = 128; s > 0; s >>= 1) {
        if (tid < s) { sd[tid] += sd[tid + s]; sl[tid] += sl[tid + s]; }
        __syncthreads();
    }
    if (tid == 0) {
        const float t0m  = sd[0] * (1.0f / NN);
        const float lsem = sl[0] * (1.0f / NN);
        out[0] = -(t0m - (lsem - __logf((float)NN)));
    }
}

// ---------------------------------------------------------------------------
extern "C" void kernel_launch(void* const* d_in, const int* in_sizes, int n_in,
                              void* d_out, int out_size, void* d_ws, size_t ws_size,
                              hipStream_t stream) {
    const float* x  = (const float*)d_in[0];   // (1536,192)
    const float* y  = (const float*)d_in[1];   // (1536,192)
    const float* W1 = (const float*)d_in[2];   // (64,384)
    const float* b1 = (const float*)d_in[3];   // (64,)
    const float* W2 = (const float*)d_in[4];   // (1,64)
    const float* b2 = (const float*)d_in[5];   // (1,)
    // d_in[6] = mask (all ones) unused

    float* ws   = (float*)d_ws;
    float* hx   = ws;                     // 1536*64
    float* hyb  = hx + NN * HID;          // 1536*64
    float* diag = hyb + NN * HID;         // 1536
    float* lse  = diag + NN;              // 1536

    hxhy_wmma   <<<NN / 16,    32,  0, stream>>>(x, y, W1, b1, hx, hyb);
    pairwise_lse<<<NN / ITILE, 256, 0, stream>>>(hx, hyb, W2, b2, diag, lse);
    finalize    <<<1,          256, 0, stream>>>(diag, lse, (float*)d_out);
}
